// SEAAttention_24764781428859
// MI455X (gfx1250) — matrix operand order repaired and checked by reading, not compiled, not run on hardware
//
#include <hip/hip_runtime.h>

// ---------------------------------------------------------------------------
// SEA axial attention for MI455X (gfx1250, wave32, WMMA bf16->f32).
// Pool-before-QKV algebraic rewrite; all GEMM-shaped work on v_wmma.
// GEMMs register-blocked 16x64 per wave: 4 independent WMMA accumulator
// chains (hides D->C hazard), A fragment reused 4x (2.5 b128 loads / wmma).
// ---------------------------------------------------------------------------

typedef __bf16 v16bf __attribute__((ext_vector_type(16)));
typedef float  v8f   __attribute__((ext_vector_type(8)));

#define BATCH 4
#define CH    256
#define HH    128
#define WW_   128
#define HEADS 4
#define DHEAD 64
#define NTOK  2048   // pooled tokens per branch (16*128 or 128*16)

__device__ __forceinline__ __bf16 f2bf(float f) {
  unsigned u = __builtin_bit_cast(unsigned, f);
  u += 0x7fffu + ((u >> 16) & 1u);               // round-to-nearest-even
  unsigned short h = (unsigned short)(u >> 16);
  return __builtin_bit_cast(__bf16, h);
}
__device__ __forceinline__ float bf2f(__bf16 h) {
  unsigned short s = __builtin_bit_cast(unsigned short, h);
  unsigned u = ((unsigned)s) << 16;
  return __builtin_bit_cast(float, u);
}

union FragU { v16bf v; uint4 q[2]; };

// A fragment: 16x32 bf16 tile, src = row-major [16][ld] tile origin.
// lane L holds row M=L%16; element j -> K = (j/8)*16 + (L/16)*8 + j%8.
__device__ __forceinline__ v16bf load_a_frag(const __bf16* src, int ld) {
  int lane = threadIdx.x & 31;
  int m = lane & 15, half = lane >> 4;
  const __bf16* base = src + m * ld + half * 8;
  FragU f;
  f.q[0] = *reinterpret_cast<const uint4*>(base);
  f.q[1] = *reinterpret_cast<const uint4*>(base + 16);
  return f.v;
}

// B fragment: 32(K)x16(N) bf16 tile, src = N-major [16][ld] tile origin.
// lane L holds col N=L%16; element j -> K = (j/8)*16 + (L/16)*8 + j%8.
__device__ __forceinline__ v16bf load_b_frag(const __bf16* src, int ld) {
  int lane = threadIdx.x & 31;
  int n = lane & 15, half = lane >> 4;
  const __bf16* base = src + n * ld + half * 8;
  FragU f;
  f.q[0] = *reinterpret_cast<const uint4*>(base);
  f.q[1] = *reinterpret_cast<const uint4*>(base + 16);
  return f.v;
}

__device__ __forceinline__ v8f wmma_bf16(v16bf a, v16bf b, v8f c) {
  return __builtin_amdgcn_wmma_f32_16x16x32_bf16(false, a, false, b,
                                                 (short)0, c, false, false);
}

// --------------------------- K0: weights -> bf16 ---------------------------
__global__ void k_cvt_weights(const float* wq, const float* wp,
                              __bf16* wqb, __bf16* wpb) {
  int i = blockIdx.x * 256 + threadIdx.x;
  if (i < 3 * CH * CH) wqb[i] = f2bf(wq[i]);
  if (i < CH * CH)     wpb[i] = f2bf(wp[i]);
}

// ----------------- K1: axial average pool, write [b][br][p][c] -------------
__global__ void k_pool(const float* __restrict__ x, __bf16* __restrict__ pooled) {
  int gid = blockIdx.x * 256 + threadIdx.x;      // 4*2*2048*256 = 2^22
  int c  = gid & 255;
  int p  = (gid >> 8) & (NTOK - 1);
  int br = (gid >> 19) & 1;
  int b  = gid >> 20;
  const float* xb = x + ((size_t)b * CH + c) * (HH * WW_);
  float s = 0.f;
  if (br == 0) {                 // pool over H: p = hh*128 + w
    int hh = p >> 7, w = p & 127;
    const float* pp = xb + (hh * 8) * WW_ + w;
#pragma unroll
    for (int d = 0; d < 8; ++d) s += pp[d * WW_];
  } else {                       // pool over W: p = h*16 + ww
    int h = p >> 4, ww = p & 15;
    const float* pp = xb + h * WW_ + ww * 8;
#pragma unroll
    for (int d = 0; d < 8; ++d) s += pp[d];
  }
  pooled[((size_t)((b * 2 + br) * NTOK + p)) * CH + c] = f2bf(s * 0.125f);
}

// -------- K2: QKV GEMM (768x256)x(256x2048) per (b,branch) via WMMA --------
// 16x64 per wave: one A fragment feeds 4 independent accumulators.
// q,k scattered as [bbh][p][d] (fragment-friendly); v as [bbh][d][p].
__global__ __launch_bounds__(256) void k_qkv(
    const __bf16* __restrict__ wqb, const float* __restrict__ bqkv,
    const __bf16* __restrict__ pooled,
    __bf16* __restrict__ qb, __bf16* __restrict__ kb, __bf16* __restrict__ vb) {
  int wid = blockIdx.x * 8 + (threadIdx.x >> 5);  // 8*48*32 wave-tiles
  int nt = wid % 32;                              // 64-wide N block
  int mt = (wid / 32) % 48;
  int bb = wid / (32 * 48);                       // b*2+br
  int lane = threadIdx.x & 31;

  const __bf16* asrc = wqb + (size_t)(mt * 16) * CH;
  const __bf16* bsrc = pooled + ((size_t)bb * NTOK + nt * 64) * CH;
  v8f acc[4] = {};
#pragma unroll
  for (int k0 = 0; k0 < CH; k0 += 32) {
    v16bf a = load_a_frag(asrc + k0, CH);
#pragma unroll
    for (int t = 0; t < 4; ++t)
      acc[t] = wmma_bf16(a, load_b_frag(bsrc + (size_t)(t * 16) * CH + k0, CH),
                         acc[t]);
  }

  int n = lane & 15, half = lane >> 4;
#pragma unroll
  for (int t = 0; t < 4; ++t) {
    int p = nt * 64 + t * 16 + n;
#pragma unroll
    for (int r = 0; r < 8; ++r) {
      int o = mt * 16 + r + half * 8;
      float val = acc[t][r] + bqkv[o];
      int sel = o >> 8, rem = o & 255, head = rem >> 6, dch = rem & 63;
      __bf16 hv = f2bf(val);
      size_t base = ((size_t)(bb * HEADS + head)) * (NTOK * DHEAD);
      if (sel == 0)      qb[base + (size_t)p * DHEAD + dch] = hv;
      else if (sel == 1) kb[base + (size_t)p * DHEAD + dch] = hv;
      else               vb[base + (size_t)dch * NTOK + p] = hv;
    }
  }
}

// ---------------- K3: flash attention, one wave = 16 query rows ------------
__global__ __launch_bounds__(256) void k_attn(
    const __bf16* __restrict__ qb, const __bf16* __restrict__ kb,
    const __bf16* __restrict__ vb, float* __restrict__ att) {
  __shared__ __bf16 pst[8][16 * 32];              // per-wave P staging (8 KB)
  int wslot = threadIdx.x >> 5;
  int wid = blockIdx.x * 8 + wslot;               // 0..4095
  int qt = wid & 127;                             // query tile
  int bbh = wid >> 7;                             // (b*2+br)*4 + head
  int head = bbh & 3;
  int bb = bbh >> 2;
  int lane = threadIdx.x & 31;
  int half = lane >> 4;

  size_t base = ((size_t)bbh) * (NTOK * DHEAD);
  const __bf16* qrow = qb + base + (size_t)(qt * 16) * DHEAD;
  const __bf16* krow = kb + base;
  const __bf16* vrow = vb + base;

  v16bf aq0 = load_a_frag(qrow, DHEAD);           // dch 0..31
  v16bf aq1 = load_a_frag(qrow + 32, DHEAD);      // dch 32..63

  float m[8], l[8];
#pragma unroll
  for (int r = 0; r < 8; ++r) { m[r] = -1e30f; l[r] = 0.f; }
  v8f acc0 = {}, acc1 = {}, acc2 = {}, acc3 = {};
  const float invscale = 1.0f / 16.0f;            // 1/sqrt(C), C=256

  for (int kt = 0; kt < NTOK; kt += 32) {
    v8f s0 = {}, s1 = {};
    s0 = wmma_bf16(aq0, load_b_frag(krow + (size_t)kt * DHEAD, DHEAD), s0);
    s1 = wmma_bf16(aq0, load_b_frag(krow + (size_t)(kt + 16) * DHEAD, DHEAD), s1);
    s0 = wmma_bf16(aq1, load_b_frag(krow + (size_t)kt * DHEAD + 32, DHEAD), s0);
    s1 = wmma_bf16(aq1, load_b_frag(krow + (size_t)(kt + 16) * DHEAD + 32, DHEAD), s1);

    float sc[8];
#pragma unroll
    for (int r = 0; r < 8; ++r) {
      float a0 = s0[r] * invscale, a1 = s1[r] * invscale;
      float mx = fmaxf(a0, a1);
#pragma unroll
      for (int sh = 1; sh < 16; sh <<= 1) mx = fmaxf(mx, __shfl_xor(mx, sh, 32));
      float mn = fmaxf(m[r], mx);
      float p0 = __expf(a0 - mn), p1 = __expf(a1 - mn);
      float rs = p0 + p1;
#pragma unroll
      for (int sh = 1; sh < 16; sh <<= 1) rs += __shfl_xor(rs, sh, 32);
      float corr = __expf(m[r] - mn);
      l[r] = l[r] * corr + rs;
      m[r] = mn;
      sc[r] = corr;
      int row = r + half * 8, col = lane & 15;
      pst[wslot][row * 32 + col]      = f2bf(p0);   // keys kt..kt+15
      pst[wslot][row * 32 + 16 + col] = f2bf(p1);   // keys kt+16..kt+31
    }
#pragma unroll
    for (int r = 0; r < 8; ++r) {
      acc0[r] *= sc[r]; acc1[r] *= sc[r]; acc2[r] *= sc[r]; acc3[r] *= sc[r];
    }
    v16bf pa = load_a_frag(&pst[wslot][0], 32);     // same-wave LDS: in-order
    acc0 = wmma_bf16(pa, load_b_frag(vrow + (size_t)0 * 16 * NTOK + kt, NTOK), acc0);
    acc1 = wmma_bf16(pa, load_b_frag(vrow + (size_t)1 * 16 * NTOK + kt, NTOK), acc1);
    acc2 = wmma_bf16(pa, load_b_frag(vrow + (size_t)2 * 16 * NTOK + kt, NTOK), acc2);
    acc3 = wmma_bf16(pa, load_b_frag(vrow + (size_t)3 * 16 * NTOK + kt, NTOK), acc3);
  }

  // write [b*2+br][c][p] f32 (channel-major for the upsample kernel)
  int n = lane & 15;
#pragma unroll
  for (int r = 0; r < 8; ++r) {
    int row = r + half * 8;
    int p = qt * 16 + row;
    float invl = 1.0f / l[r];
    size_t cb = ((size_t)bb * CH + head * DHEAD + n) * NTOK + p;
    att[cb + (size_t)0  * NTOK] = acc0[r] * invl;
    att[cb + (size_t)16 * NTOK] = acc1[r] * invl;
    att[cb + (size_t)32 * NTOK] = acc2[r] * invl;
    att[cb + (size_t)48 * NTOK] = acc3[r] * invl;
  }
}

// ------------- K4a: bilinear upsample both branches + sum -> bf16 ----------
__global__ void k_upsample(const float* __restrict__ att, __bf16* __restrict__ comb) {
  int gid = blockIdx.x * 256 + threadIdx.x;       // 2^24, layout [b][c][h][w]
  int w = gid & 127, h = (gid >> 7) & 127;
  int c = (gid >> 14) & 255;
  int b = gid >> 22;
  // H branch (16 -> 128, align_corners=False)
  float sh = fminf(fmaxf((h + 0.5f) * 0.125f - 0.5f, 0.f), 15.f);
  int h0 = (int)sh; int h1 = h0 + 1 < 15 ? h0 + 1 : 15;
  float w1 = sh - (float)h0, w0 = 1.f - w1;
  const float* ah = att + ((size_t)((b * 2 + 0) * CH + c)) * NTOK;
  float vh = ah[h0 * 128 + w] * w0 + ah[h1 * 128 + w] * w1;
  // W branch
  float sw = fminf(fmaxf((w + 0.5f) * 0.125f - 0.5f, 0.f), 15.f);
  int x0 = (int)sw; int x1 = x0 + 1 < 15 ? x0 + 1 : 15;
  float u1 = sw - (float)x0, u0 = 1.f - u1;
  const float* aw = att + ((size_t)((b * 2 + 1) * CH + c)) * NTOK;
  float vw = aw[h * 16 + x0] * u0 + aw[h * 16 + x1] * u1;
  comb[gid] = f2bf(vh + vw);
}

// ---------- K4b: depthwise 3x3 (+bias), write y as [P][c] bf16 -------------
__global__ void k_dwconv(const __bf16* __restrict__ comb,
                         const float* __restrict__ wdw,
                         const float* __restrict__ bdw,
                         __bf16* __restrict__ y) {
  int gid = blockIdx.x * 256 + threadIdx.x;
  int w = gid & 127, h = (gid >> 7) & 127;
  int c = (gid >> 14) & 255;
  int b = gid >> 22;
  const __bf16* src = comb + ((size_t)(b * CH + c)) * (HH * WW_);
  const float* wk = wdw + c * 9;
  float s = bdw[c];
#pragma unroll
  for (int ky = 0; ky < 3; ++ky) {
    int hy = h + ky - 1;
    if (hy < 0 || hy > 127) continue;
#pragma unroll
    for (int kx = 0; kx < 3; ++kx) {
      int wx = w + kx - 1;
      if (wx < 0 || wx > 127) continue;
      s += bf2f(src[hy * 128 + wx]) * wk[ky * 3 + kx];
    }
  }
  size_t P = (size_t)b * (HH * WW_) + h * 128 + w;
  y[P * CH + c] = f2bf(s);
}

// ---------------- K5: 1x1 projection GEMM (WMMA) + bias -> f32 -------------
// 16x64 per wave: one A fragment feeds 4 independent accumulators.
__global__ __launch_bounds__(256) void k_proj(
    const __bf16* __restrict__ wpb, const float* __restrict__ bproj,
    const __bf16* __restrict__ y, float* __restrict__ out) {
  int wid = blockIdx.x * 8 + (threadIdx.x >> 5);  // 16*1024 wave-tiles
  int nt = wid & 1023;                            // 64-wide N block
  int mt = wid >> 10;
  int lane = threadIdx.x & 31;
  const __bf16* asrc = wpb + (size_t)(mt * 16) * CH;
  const __bf16* bsrc = y + (size_t)(nt * 64) * CH;
  v8f acc[4] = {};
#pragma unroll
  for (int k0 = 0; k0 < CH; k0 += 32) {
    v16bf a = load_a_frag(asrc + k0, CH);
#pragma unroll
    for (int t = 0; t < 4; ++t)
      acc[t] = wmma_bf16(a, load_b_frag(bsrc + (size_t)(t * 16) * CH + k0, CH),
                         acc[t]);
  }
  int n = lane & 15, half = lane >> 4;
#pragma unroll
  for (int t = 0; t < 4; ++t) {
    int P = nt * 64 + t * 16 + n;
    int b = P >> 14, hw = P & 16383;
#pragma unroll
    for (int r = 0; r < 8; ++r) {
      int o = mt * 16 + r + half * 8;
      out[((size_t)b * CH + o) * (HH * WW_) + hw] = acc[t][r] + bproj[o];
    }
  }
}

// ---------------------------------------------------------------------------
extern "C" void kernel_launch(void* const* d_in, const int* in_sizes, int n_in,
                              void* d_out, int out_size, void* d_ws, size_t ws_size,
                              hipStream_t stream) {
  (void)in_sizes; (void)n_in; (void)out_size; (void)ws_size;
  const float* x      = (const float*)d_in[0];
  const float* w_qkv  = (const float*)d_in[1];
  const float* b_qkv  = (const float*)d_in[2];
  const float* w_dw   = (const float*)d_in[3];
  const float* b_dw   = (const float*)d_in[4];
  const float* w_proj = (const float*)d_in[5];
  const float* b_proj = (const float*)d_in[6];
  float* out = (float*)d_out;

  char* ws = (char*)d_ws;
  const size_t MB = 1024 * 1024;
  // Live-range packed layout (~80.5 MB total):
  __bf16* pooled = (__bf16*)(ws + 0);        // [ 0, 8) MB, dead after K2
  __bf16* qbuf   = (__bf16*)(ws + 8  * MB);  // [ 8,16) MB, dead after K3
  __bf16* kbuf   = (__bf16*)(ws + 16 * MB);  // [16,24) MB, dead after K3
  __bf16* vbuf   = (__bf16*)(ws + 24 * MB);  // [24,32) MB, dead after K3
  float*  att    = (float*)(ws + 32 * MB);   // [32,48) MB, dead after K4a
  __bf16* comb   = (__bf16*)(ws + 0);        // [ 0,32) MB, reuses pooled/q/k/v
  __bf16* ybuf   = (__bf16*)(ws + 48 * MB);  // [48,80) MB
  __bf16* wqb    = (__bf16*)(ws + 80 * MB);
  __bf16* wpb    = (__bf16*)(ws + 80 * MB + (size_t)3 * CH * CH * 2);

  k_cvt_weights<<<(3 * CH * CH + 255) / 256, 256, 0, stream>>>(w_qkv, w_proj, wqb, wpb);
  k_pool<<<(BATCH * 2 * NTOK * CH) / 256, 256, 0, stream>>>(x, pooled);
  k_qkv<<<(8 * 48 * 32) / 8, 256, 0, stream>>>(wqb, b_qkv, pooled, qbuf, kbuf, vbuf);
  k_attn<<<(32 * 128) / 8, 256, 0, stream>>>(qbuf, kbuf, vbuf, att);
  k_upsample<<<(BATCH * CH * HH * WW_) / 256, 256, 0, stream>>>(att, comb);
  k_dwconv<<<(BATCH * CH * HH * WW_) / 256, 256, 0, stream>>>(comb, w_dw, b_dw, ybuf);
  k_proj<<<(16 * 1024) / 8, 256, 0, stream>>>(wpb, b_proj, ybuf, out);
}
